// SelfAttention_86406152061076
// MI455X (gfx1250) — matrix operand order repaired and checked by reading, not compiled, not run on hardware
//
#include <hip/hip_runtime.h>
#include <cstdint>
#include <cstddef>

// ---------- types ----------
typedef __attribute__((ext_vector_type(16))) __bf16   bf16x16;
typedef __attribute__((ext_vector_type(8)))  __bf16   bf16x8;
typedef __attribute__((ext_vector_type(8)))  float    floatx8;
typedef __attribute__((ext_vector_type(4)))  float    floatx4;
typedef __attribute__((ext_vector_type(4)))  uint32_t uint32x4;
typedef __attribute__((ext_vector_type(8)))  int      int32x8;
typedef __attribute__((ext_vector_type(4)))  int      int32x4;

#if defined(__has_builtin)
#if __has_builtin(__builtin_amdgcn_tensor_load_to_lds)
#define USE_TDM 1
#else
#define USE_TDM 0
#endif
#else
#define USE_TDM 0
#endif

// ---------- bf16 <-> f32 helpers (RNE) ----------
__device__ inline __bf16 f2bf(float f) {
    union { float f; uint32_t u; } a; a.f = f;
    uint32_t r = a.u + 0x7FFFu + ((a.u >> 16) & 1u);
    uint16_t hv = (uint16_t)(r >> 16);
    __bf16 out;
    __builtin_memcpy(&out, &hv, 2);
    return out;
}
__device__ inline float bf2f(__bf16 b) {
    uint16_t hv; __builtin_memcpy(&hv, &b, 2);
    union { uint32_t u; float f; } a; a.u = ((uint32_t)hv) << 16;
    return a.f;
}

__device__ inline bf16x16 cat16(bf16x8 u0, bf16x8 u1) {
    bf16x16 r;
#pragma unroll
    for (int i = 0; i < 8; ++i) { r[i] = u0[i]; r[i + 8] = u1[i]; }
    return r;
}

// LDS tile stride: 40 bf16 = 80 B -> rows 16B-aligned, 20-bank skew so the
// half-wave fragment gathers (b128) are bank-conflict-free.
#define LSTR 40

// flat pointer low 32 bits == LDS byte offset
__device__ inline uint32_t lds_addr32(const void* p) {
    return (uint32_t)(uintptr_t)p;
}

#if USE_TDM
// ---------------------------------------------------------------------
// Tensor Data Mover: load a 128-row x 32-col bf16 tile from a row-major
// [rows x K] bf16 tensor into LDS with an extra 16 B of pad per 64 B row
// (pad_interval=16 DW -> code 3, pad_amount=4 DW -> code 3), i.e. exactly
// the LSTR=40-element padded layout the WMMA fragment gathers expect.
// D# bit layout per CDNA5 ISA section 8.3/8.4.
// ---------------------------------------------------------------------
__device__ inline void tdm_load_tile128x32(const __bf16* gsrc, uint32_t lds_off,
                                           uint32_t K, uint32_t rows) {
    uint64_t ga = (uint64_t)(uintptr_t)gsrc;
    uint32x4 g0;
    g0[0] = 1u;                                              // count=1, user mode
    g0[1] = lds_off;                                         // lds_addr
    g0[2] = (uint32_t)(ga & 0xFFFFFFFFu);                    // global_addr[31:0]
    g0[3] = (uint32_t)((ga >> 32) & 0x01FFFFFFu) | (2u << 30); // addr[56:32], type=2
    int32x8 g1;
    g1[0] = (int)((1u << 16) |                               // data_size = 2 B
                  (1u << 20) |                               // pad_enable
                  (3u << 22) |                               // pad_interval: 16 DWORDs
                  (3u << 25));                               // pad_amount:   4 DWORDs
    g1[1] = (int)((K & 0xFFFFu) << 16);                      // tensor_dim0 lo16
    g1[2] = (int)(((K >> 16) & 0xFFFFu)) | (int)((rows & 0xFFFFu) << 16); // dim0 hi, dim1 lo
    g1[3] = (int)(((rows >> 16) & 0xFFFFu)) | (int)(32u << 16); // dim1 hi, tile_dim0=32
    g1[4] = 128;                                             // tile_dim1 = 128 rows
    g1[5] = (int)K;                                          // tensor_dim0_stride lo32
    g1[6] = 0;                                               // stride hi, dim1_stride lo
    g1[7] = 0;
    int32x4 z4 = {0, 0, 0, 0};
#if __clang_major__ >= 23
    int32x8 z8 = {0, 0, 0, 0, 0, 0, 0, 0};
    __builtin_amdgcn_tensor_load_to_lds(g0, g1, z4, z4, z8, 0);
#else
    __builtin_amdgcn_tensor_load_to_lds(g0, g1, z4, z4, 0);
#endif
}
#endif

// =====================================================================
// fp32 -> bf16 one-shot conversion (vectorized), run once per operand so
// GEMMs stream pure bf16 (half the L2/HBM traffic of repeated f32 reads)
// =====================================================================
__global__ __launch_bounds__(256) void f32_to_bf16_kernel(
    const float* __restrict__ in, __bf16* __restrict__ out, int n8)
{
    int i = blockIdx.x * blockDim.x + threadIdx.x;
    if (i < n8) {
        const floatx4* p4 = (const floatx4*)(in + (size_t)i * 8);
        floatx4 a = p4[0], b = p4[1];
        bf16x8 r;
        r[0] = f2bf(a[0]); r[1] = f2bf(a[1]); r[2] = f2bf(a[2]); r[3] = f2bf(a[3]);
        r[4] = f2bf(b[0]); r[5] = f2bf(b[1]); r[6] = f2bf(b[2]); r[7] = f2bf(b[3]);
        *(bf16x8*)(out + (size_t)i * 8) = r;
    }
}

// =====================================================================
// bf16 GEMM: C[M,N] = A[M,K] * B[N,K]^T, fp32 accumulate.
// Block: 256 threads (8 waves), tile 128(M) x 128(N), K step 32.
// Waves: 4(M) x 2(N); wave tile 32x64 = 2x4 wmma accumulators.
// Tiles double-buffered in LDS; loaded by TDM (wave 0 issues, all waves
// sync via s_wait_tensorcnt + barrier) or manual b128 loads as fallback.
// EPI 0: fp32 store (ldc=N).  EPI 1: scatter qkv -> q/k/v [B,H,N,hd] bf16.
// =====================================================================
template <int EPI>
__global__ __launch_bounds__(256) void gemm_bf16_kernel(
    const __bf16* __restrict__ A, const __bf16* __restrict__ B, float* __restrict__ Cout,
    __bf16* __restrict__ qb, __bf16* __restrict__ kb, __bf16* __restrict__ vb,
    int M, int N, int K)
{
    __shared__ __align__(16) __bf16 As[2][128 * LSTR];
    __shared__ __align__(16) __bf16 Bs[2][128 * LSTR];

    const int tid  = threadIdx.x;
    const int bn   = blockIdx.x;   // N tile (128)
    const int bm   = blockIdx.y;   // M tile (128)
    const int wave = tid >> 5;
    const int lane = tid & 31;
    const int lm   = lane & 15;
    const int g    = lane >> 4;
    const int wm   = wave & 3;     // 4 waves in M
    const int wn   = wave >> 2;    // 2 waves in N

    floatx8 acc[2][4];
#pragma unroll
    for (int ms = 0; ms < 2; ++ms)
#pragma unroll
        for (int ns = 0; ns < 4; ++ns)
#pragma unroll
            for (int r = 0; r < 8; ++r) acc[ms][ns][r] = 0.0f;

    const int ntiles = K >> 5;

#if USE_TDM
    if (wave == 0) {  // prologue: fill buffer 0
        tdm_load_tile128x32(A + (size_t)bm * 128 * K, lds_addr32(&As[0][0]), (uint32_t)K, (uint32_t)M);
        tdm_load_tile128x32(B + (size_t)bn * 128 * K, lds_addr32(&Bs[0][0]), (uint32_t)K, (uint32_t)N);
    }
#endif

    for (int kt = 0; kt < ntiles; ++kt) {
        const int cur = kt & 1;
#if USE_TDM
        if (wave == 0) {
            if (kt + 1 < ntiles) {  // issue next tile into other buffer, then wait for current
                tdm_load_tile128x32(A + (size_t)bm * 128 * K + (kt + 1) * 32,
                                    lds_addr32(&As[cur ^ 1][0]), (uint32_t)K, (uint32_t)M);
                tdm_load_tile128x32(B + (size_t)bn * 128 * K + (kt + 1) * 32,
                                    lds_addr32(&Bs[cur ^ 1][0]), (uint32_t)K, (uint32_t)N);
                __builtin_amdgcn_s_wait_tensorcnt((short)2);
            } else {
                __builtin_amdgcn_s_wait_tensorcnt((short)0);
            }
        }
        __syncthreads();
#else
        // manual fallback: A,B tiles 128x32 = 512 chunks of 8 each
#pragma unroll
        for (int i = 0; i < 2; ++i) {
            int ch = tid * 2 + i, row = ch >> 2, cc = ch & 3;
            *(bf16x8*)&As[cur][row * LSTR + cc * 8] =
                *(const bf16x8*)(A + (size_t)(bm * 128 + row) * K + kt * 32 + cc * 8);
            *(bf16x8*)&Bs[cur][row * LSTR + cc * 8] =
                *(const bf16x8*)(B + (size_t)(bn * 128 + row) * K + kt * 32 + cc * 8);
        }
        __syncthreads();
#endif
        // fragment gathers per documented 16-bit WMMA layouts
        bf16x16 aF[2], bF[4];
#pragma unroll
        for (int ms = 0; ms < 2; ++ms) {
            int m = wm * 32 + ms * 16 + lm;  // A: lane=row, K runs g*8.. and 16+g*8..
            aF[ms] = cat16(*(const bf16x8*)&As[cur][m * LSTR + g * 8],
                           *(const bf16x8*)&As[cur][m * LSTR + 16 + g * 8]);
        }
#pragma unroll
        for (int ns = 0; ns < 4; ++ns) {
            int nc = wn * 64 + ns * 16 + lm; // B: lane=col, K runs g*8.. and 16+g*8..
            bF[ns] = cat16(*(const bf16x8*)&Bs[cur][nc * LSTR + g * 8],
                           *(const bf16x8*)&Bs[cur][nc * LSTR + 16 + g * 8]);
        }
#pragma unroll
        for (int ms = 0; ms < 2; ++ms)
#pragma unroll
            for (int ns = 0; ns < 4; ++ns)
                acc[ms][ns] = __builtin_amdgcn_wmma_f32_16x16x32_bf16(
                    false, aF[ms], false, bF[ns], (short)0, acc[ms][ns], false, false);
        __syncthreads();
    }

    // epilogue (C layout: lane holds N=lm, M = r + 8*g per VGPR r)
#pragma unroll
    for (int ms = 0; ms < 2; ++ms)
#pragma unroll
        for (int ns = 0; ns < 4; ++ns)
#pragma unroll
            for (int r = 0; r < 8; ++r) {
                int row = bm * 128 + wm * 32 + ms * 16 + r + 8 * g;  // token t
                int col = bn * 128 + wn * 64 + ns * 16 + lm;         // out dim d
                float val = acc[ms][ns][r];
                if (EPI == 0) {
                    Cout[(size_t)row * N + col] = val;
                } else {
                    int which = col >> 10, rem = col & 1023;
                    int h = rem >> 6, e = rem & 63;
                    int b = row >> 10, n = row & 1023;
                    size_t o = ((size_t)((b * 16 + h) * 1024 + n)) * 64 + e;
                    __bf16 bv = f2bf(val);
                    if (which == 0)      qb[o] = bv;
                    else if (which == 1) kb[o] = bv;
                    else                 vb[o] = bv;
                }
            }
}

// =====================================================================
// RoPE in-place on q (fused * hd^-0.5) and k; one thread per (b,h,n,pair)
// =====================================================================
__global__ __launch_bounds__(256) void rope_kernel(
    __bf16* __restrict__ qb, __bf16* __restrict__ kb,
    const float* __restrict__ pos, float scale)
{
    int idx = blockIdx.x * blockDim.x + threadIdx.x;  // B*H*N*32
    int e  = idx & 31;
    int n  = (idx >> 5) & 1023;
    int bh = idx >> 15;
    size_t base = ((size_t)bh * 1024 + n) * 64;
    float c1 = __cosf(pos[n * 64 + e]),      s1 = __sinf(pos[n * 64 + e]);
    float c2 = __cosf(pos[n * 64 + e + 32]), s2 = __sinf(pos[n * 64 + e + 32]);
    float q1 = bf2f(qb[base + e]), q2 = bf2f(qb[base + e + 32]);
    float k1 = bf2f(kb[base + e]), k2 = bf2f(kb[base + e + 32]);
    qb[base + e]      = f2bf((q1 * c1 - q2 * s1) * scale);
    qb[base + e + 32] = f2bf((q2 * c2 + q1 * s2) * scale);
    kb[base + e]      = f2bf(k1 * c1 - k2 * s1);
    kb[base + e + 32] = f2bf(k2 * c2 + k1 * s2);
}

// =====================================================================
// Flash attention: block = 64 query rows of one (b,h); 4 waves x 16 rows.
// =====================================================================
__global__ __launch_bounds__(128) void attn_kernel(
    const __bf16* __restrict__ q, const __bf16* __restrict__ k,
    const __bf16* __restrict__ v, const unsigned char* __restrict__ mask,
    __bf16* __restrict__ o)
{
    __shared__ __align__(16) __bf16 Kt[64 * LSTR];       // [key][dim]
    __shared__ __align__(16) __bf16 Vt[64 * LSTR];       // transposed: [dim][key]
    __shared__ __align__(16) __bf16 Ps[4 * 16 * LSTR];   // per-wave P

    const int qt   = blockIdx.x;
    const int h    = blockIdx.y;
    const int b    = blockIdx.z;
    const int tid  = threadIdx.x;
    const int wave = tid >> 5, lane = tid & 31, lm = lane & 15, g = lane >> 4;

    const size_t head = (size_t)(b * 16 + h) * 1024;

    bf16x16 qf[2];
    {
        const __bf16* qr = q + (head + qt * 64 + wave * 16 + lm) * 64;
#pragma unroll
        for (int ks = 0; ks < 2; ++ks)
            qf[ks] = cat16(*(const bf16x8*)(qr + ks * 32 + g * 8),
                           *(const bf16x8*)(qr + ks * 32 + 16 + g * 8));
    }

    floatx8 m_run, l_run, accO[4];
#pragma unroll
    for (int r = 0; r < 8; ++r) { m_run[r] = -3.0e38f; l_run[r] = 0.0f; }
#pragma unroll
    for (int no = 0; no < 4; ++no)
#pragma unroll
        for (int r = 0; r < 8; ++r) accO[no][r] = 0.0f;

    __bf16* psw = &Ps[wave * 16 * LSTR];

    for (int j = 0; j < 16; ++j) {
        // prefetch next K/V tiles (streamed once; gfx1250 global_prefetch_b8)
        if (j + 1 < 16) {
            __builtin_prefetch(k + (head + (j + 1) * 64 + (tid >> 1)) * 64, 0, 1);
            __builtin_prefetch(v + (head + (j + 1) * 64 + (tid >> 1)) * 64, 0, 1);
        }
        // K tile, row-major
#pragma unroll
        for (int i = 0; i < 4; ++i) {
            int ch = tid * 4 + i, row = ch >> 3, cc = ch & 7;
            *(bf16x8*)&Kt[row * LSTR + cc * 8] =
                *(const bf16x8*)(k + (head + j * 64 + row) * 64 + cc * 8);
        }
        // V tile, transposed into [dim][key]
        {
            int key = tid >> 1, d0 = (tid & 1) * 32;
            const __bf16* vp = v + (head + j * 64 + key) * 64 + d0;
#pragma unroll
            for (int u = 0; u < 4; ++u) {
                bf16x8 val = *(const bf16x8*)(vp + u * 8);
#pragma unroll
                for (int m2 = 0; m2 < 8; ++m2)
                    Vt[(d0 + u * 8 + m2) * LSTR + key] = val[m2];
            }
        }
        __syncthreads();

        // S = Q K^T
        floatx8 s[4];
#pragma unroll
        for (int ns = 0; ns < 4; ++ns) {
#pragma unroll
            for (int r = 0; r < 8; ++r) s[ns][r] = 0.0f;
            int key = ns * 16 + lm;
#pragma unroll
            for (int ks = 0; ks < 2; ++ks) {
                bf16x16 bf = cat16(*(const bf16x8*)&Kt[key * LSTR + ks * 32 + g * 8],
                                   *(const bf16x8*)&Kt[key * LSTR + ks * 32 + 16 + g * 8]);
                s[ns] = __builtin_amdgcn_wmma_f32_16x16x32_bf16(
                    false, qf[ks], false, bf, (short)0, s[ns], false, false);
            }
        }

        // key padding mask
#pragma unroll
        for (int ns = 0; ns < 4; ++ns) {
            int kc = j * 64 + ns * 16 + lm;
            if (!mask[b * 1024 + kc]) {
#pragma unroll
                for (int r = 0; r < 8; ++r) s[ns][r] = -3.0e38f;
            }
        }

        // online softmax; row r+8g lives across the 16 lanes of one half
        floatx8 mnew;
#pragma unroll
        for (int r = 0; r < 8; ++r) {
            float mx = fmaxf(fmaxf(s[0][r], s[1][r]), fmaxf(s[2][r], s[3][r]));
            mx = fmaxf(mx, __shfl_xor(mx, 1));
            mx = fmaxf(mx, __shfl_xor(mx, 2));
            mx = fmaxf(mx, __shfl_xor(mx, 4));
            mx = fmaxf(mx, __shfl_xor(mx, 8));
            mnew[r] = fmaxf(m_run[r], mx);
        }
#pragma unroll
        for (int r = 0; r < 8; ++r) {
            float alpha = __expf(m_run[r] - mnew[r]);
            l_run[r] *= alpha;
#pragma unroll
            for (int no = 0; no < 4; ++no) accO[no][r] *= alpha;
            m_run[r] = mnew[r];
        }
#pragma unroll
        for (int ns = 0; ns < 4; ++ns)
#pragma unroll
            for (int r = 0; r < 8; ++r) s[ns][r] = __expf(s[ns][r] - mnew[r]);
#pragma unroll
        for (int r = 0; r < 8; ++r) {
            float rs = s[0][r] + s[1][r] + s[2][r] + s[3][r];
            rs += __shfl_xor(rs, 1);
            rs += __shfl_xor(rs, 2);
            rs += __shfl_xor(rs, 4);
            rs += __shfl_xor(rs, 8);
            l_run[r] += rs;
        }

        // stage P (C-layout -> A-layout) through per-wave LDS
#pragma unroll
        for (int ns = 0; ns < 4; ++ns)
#pragma unroll
            for (int r = 0; r < 8; ++r)
                psw[(r + 8 * g) * LSTR + ns * 16 + lm] = f2bf(s[ns][r]);

        // O += P V
#pragma unroll
        for (int ks2 = 0; ks2 < 2; ++ks2) {
            bf16x16 pf = cat16(*(const bf16x8*)&psw[lm * LSTR + ks2 * 32 + g * 8],
                               *(const bf16x8*)&psw[lm * LSTR + ks2 * 32 + 16 + g * 8]);
#pragma unroll
            for (int no = 0; no < 4; ++no) {
                int dc = no * 16 + lm;
                bf16x16 vf = cat16(*(const bf16x8*)&Vt[dc * LSTR + ks2 * 32 + g * 8],
                                   *(const bf16x8*)&Vt[dc * LSTR + ks2 * 32 + 16 + g * 8]);
                accO[no] = __builtin_amdgcn_wmma_f32_16x16x32_bf16(
                    false, pf, false, vf, (short)0, accO[no], false, false);
            }
        }
        __syncthreads();
    }

    // finalize
#pragma unroll
    for (int no = 0; no < 4; ++no)
#pragma unroll
        for (int r = 0; r < 8; ++r) {
            float inv = 1.0f / l_run[r];
            int m  = r + 8 * g;
            int nq = qt * 64 + wave * 16 + m;
            int c  = h * 64 + no * 16 + lm;
            o[((size_t)(b * 1024 + nq)) * 1024 + c] = f2bf(accO[no][r] * inv);
        }
}

// =====================================================================
// launch
// =====================================================================
extern "C" void kernel_launch(void* const* d_in, const int* in_sizes, int n_in,
                              void* d_out, int out_size, void* d_ws, size_t ws_size,
                              hipStream_t stream)
{
    const float*         x     = (const float*)d_in[0];
    const unsigned char* mask  = (const unsigned char*)d_in[1];
    const float*         pos   = (const float*)d_in[2];
    const float*         wqkv  = (const float*)d_in[3];
    const float*         wproj = (const float*)d_in[4];
    float*               out   = (float*)d_out;

    char* ws = (char*)d_ws;
    const size_t SZ = (size_t)4 * 16 * 1024 * 64 * 2;  // 8 MB
    __bf16* qbuf = (__bf16*)(ws);
    __bf16* kbuf = (__bf16*)(ws + SZ);
    __bf16* vbuf = (__bf16*)(ws + 2 * SZ);
    __bf16* obuf = (__bf16*)(ws + 3 * SZ);
    __bf16* xb   = (__bf16*)(ws + 4 * SZ);             // 8 MB
    __bf16* wqb  = (__bf16*)(ws + 5 * SZ);             // 6 MB
    __bf16* wpb  = (__bf16*)(ws + 5 * SZ + 6291456);   // 2 MB   (total 48 MB)

    // 0) one-shot bf16 conversion of GEMM operands
    f32_to_bf16_kernel<<<(4194304 / 8) / 256, 256, 0, stream>>>(x,     xb,  4194304 / 8);
    f32_to_bf16_kernel<<<(3145728 / 8) / 256, 256, 0, stream>>>(wqkv,  wqb, 3145728 / 8);
    f32_to_bf16_kernel<<<(1048576 / 8) / 256, 256, 0, stream>>>(wproj, wpb, 1048576 / 8);

    // 1) QKV projection: [4096,1024] x [3072,1024]^T, scatter into q/k/v
    gemm_bf16_kernel<1><<<dim3(3072 / 128, 4096 / 128), 256, 0, stream>>>(
        xb, wqb, nullptr, qbuf, kbuf, vbuf, 4096, 3072, 1024);

    // 2) RoPE in place on q (fused * hd^-0.5) and k
    rope_kernel<<<(4 * 16 * 1024 * 32) / 256, 256, 0, stream>>>(qbuf, kbuf, pos, 0.125f);

    // 3) flash attention
    attn_kernel<<<dim3(16, 16, 4), 128, 0, stream>>>(qbuf, kbuf, vbuf, mask, obuf);

    // 4) output projection -> fp32 out
    gemm_bf16_kernel<0><<<dim3(1024 / 128, 4096 / 128), 256, 0, stream>>>(
        obuf, wpb, out, nullptr, nullptr, nullptr, 4096, 1024, 1024);
}
